// FocalSparseBEVBackBoneMedian_37958920962321
// MI455X (gfx1250) — matrix operand order repaired and checked by reading, not compiled, not run on hardware
//
#include <hip/hip_runtime.h>
#include <hip/hip_bf16.h>
#include <cstdint>
#include <cstddef>

// =====================================================================
// FocalSparseBEVBackBone for MI455X (gfx1250), wave32 + WMMA bf16 path.
// Round 3 -> 4 change: fix async builtin argument types —
// __builtin_amdgcn_global_load_async_to_lds_b128 expects
// (int4 AS1*, int4 AS3*, imm, imm) per the hipcc diagnostic.
// Workspace: ~206 MiB, see offsets in kernel_launch.
// =====================================================================

using u8  = unsigned char;
using u16 = unsigned short;
using u32 = unsigned int;
using i64 = long long;

typedef __attribute__((ext_vector_type(16))) __bf16 v16bf;
typedef __attribute__((ext_vector_type(8)))  float  v8f;
typedef __attribute__((ext_vector_type(4)))  u32    u32x4;
typedef __attribute__((ext_vector_type(8)))  u32    u32x8;

__device__ __forceinline__ u16 f2bf(float f) {
  u32 x = __builtin_bit_cast(u32, f);
  return (u16)((x + 0x7fffu + ((x >> 16) & 1u)) >> 16);   // RNE
}
__device__ __forceinline__ float bf2f(u16 h) {
  u32 x = ((u32)h) << 16;
  return __builtin_bit_cast(float, x);
}

enum { F_BN = 1, F_RELU = 2, F_RES = 4, F_MASK = 8, F_OUTF = 16 };

// ---- CDNA5 async global->LDS support (probe-guarded) ----------------
#if defined(__HIP_DEVICE_COMPILE__) && __has_builtin(__builtin_amdgcn_global_load_async_to_lds_b128)
#define HAS_ASYNC_LDS 1
#else
#define HAS_ASYNC_LDS 0
#endif

#if HAS_ASYNC_LDS
typedef int v4i_async __attribute__((vector_size(4 * sizeof(int))));
typedef __attribute__((address_space(1))) v4i_async* g_v4i_ptr;
typedef __attribute__((address_space(3))) v4i_async* l_v4i_ptr;
__device__ __forceinline__ void async_ld16(const u16* g, u16* l) {
  __builtin_amdgcn_global_load_async_to_lds_b128(
      (g_v4i_ptr)(g), (l_v4i_ptr)(l), 0, 0);
}
template <int N> __device__ __forceinline__ void wait_async() {
#if __has_builtin(__builtin_amdgcn_s_wait_asynccnt)
  __builtin_amdgcn_s_wait_asynccnt(N);
#else
  asm volatile("s_wait_asynccnt %0" :: "i"(N) : "memory");
#endif
}
#endif

// ---------------------------------------------------------------------
// Scatter voxel features (atomicAdd) into dense fp32 grid [16,256,256,8]
// (channels 5..7 stay zero) and set the active-site mask.
// ---------------------------------------------------------------------
__global__ void k_scatter(const float* __restrict__ vf, const int* __restrict__ vc,
                          int N, float* __restrict__ grid, float* __restrict__ mask) {
  int i = blockIdx.x * blockDim.x + threadIdx.x;
  if (i >= N) return;
  int z = vc[i * 4 + 1], y = vc[i * 4 + 2], x = vc[i * 4 + 3];
  u32 cell = ((u32)z * 256u + (u32)y) * 256u + (u32)x;
  #pragma unroll
  for (int c = 0; c < 5; ++c) atomicAdd(&grid[(size_t)cell * 8 + c], vf[i * 5 + c]);
  mask[cell] = 1.0f;
}

__global__ void k_cvt_f32_bf16(const float* __restrict__ src, u16* __restrict__ dst, int n) {
  int i = blockIdx.x * blockDim.x + threadIdx.x;
  if (i < n) dst[i] = f2bf(src[i]);
}

// ---------------------------------------------------------------------
// Pack fp32 DHWIO (or HWIO) weights into the WMMA B-fragment lane layout:
//   index = (((tap*KC + kc)*NC + nc)*32 + lane)*16 + h
//   lane: N = nc*16 + (lane&15);  K = kc*32 + (lane>>4)*16 + h
// Zero-fills K >= CinTrue (channel padding).
// ---------------------------------------------------------------------
__global__ void k_pack_w(const float* __restrict__ w, u16* __restrict__ wpk,
                         int KT, int KC, int NC, int CinTrue, int Cout) {
  int id = blockIdx.x * blockDim.x + threadIdx.x;
  int total = KT * KC * NC * 32;
  if (id >= total) return;
  int lane = id & 31; int t = id >> 5;
  int nc = t % NC; t /= NC;
  int kc = t % KC; int tap = t / KC;
  int N = nc * 16 + (lane & 15);
  int kbase = kc * 32 + (lane >> 4) * 16;
  u16* dst = wpk + (size_t)id * 16;
  #pragma unroll
  for (int h = 0; h < 16; ++h) {
    int k = kbase + h;
    float f = (k < CinTrue) ? w[((size_t)tap * CinTrue + k) * Cout + N] : 0.0f;
    dst[h] = f2bf(f);
  }
}

// ---------------------------------------------------------------------
// WMMA implicit-GEMM conv. Each wave: 16 output x-positions x NPW*16 couts.
// A fragment (16x32 bf16, k = 16*(lane>>4) + s): per-lane two contiguous 16B channel
// chunks -> either direct global b128 loads, or async global->LDS double
// buffering when the toolchain exposes the gfx1250 async builtins.
// Out-of-range taps / padded channels read from a zeroed scratch page so
// the pipeline needs no divergence and EXEC stays all-ones for WMMA.
// ---------------------------------------------------------------------
struct ConvArgs {
  const u16* in; const u16* wpk; const u16* zp;   // zp: 4KiB zero page
  const float* bias; const float* gam; const float* bet;
  const u16* resid; const float* mask;
  u16* out; float* outf;
  int D, H, W, OD, OH, OW;
  int sz, sy, sx;
  int Cs, flags;
  int lgNC, lgNCG, lgXT, lgOH;    // log2: cout chunks, cout groups, x tiles, OH
  int totalWaves;
};

template <int KC, int KZ, int NPW>
__global__ void __launch_bounds__(128) k_conv_wmma(ConvArgs a) {
  constexpr int lgKC = (KC == 1) ? 0 : (KC == 2) ? 1 : 2;
  constexpr int ITERS = KZ * 9 * KC;
  const int lane = threadIdx.x & 31;
  const int wave = threadIdx.x >> 5;
  int wid = blockIdx.x * 4 + wave;
  if (wid >= a.totalWaves) return;              // wave-uniform exit (EXEC all-ones)
  const int ncg = wid & ((1 << a.lgNCG) - 1);
  int t = wid >> a.lgNCG;
  const int xt = (t & ((1 << a.lgXT) - 1)) << 4;
  t >>= a.lgXT;
  const int oy = t & ((1 << a.lgOH) - 1);
  const int oz = t >> a.lgOH;

  const int M    = lane & 15;
  const int hsel = lane >> 4;
  const int ox   = xt + M;
  const bool rowv = ox < a.OW;
  const int lgCout = a.lgNC + 4;

  // per-iteration A source pointers (two 16B chunks per lane)
  auto srcPtrs = [&](int it, const u16*& g0, const u16*& g1) {
    int tap = it >> lgKC;
    int kc  = it & (KC - 1);
    int dz = (KZ == 3) ? (tap / 9) : 1;
    int dy = (tap / 3) % 3, dx = tap % 3;
    int iz = oz * a.sz + dz - 1;
    int iy = oy * a.sy + dy - 1;
    bool tv = ((unsigned)iz < (unsigned)a.D) && ((unsigned)iy < (unsigned)a.H);
    int ix = ox * a.sx + dx - 1;
    bool xv = tv && rowv && ((unsigned)ix < (unsigned)a.W);
    const u16* base = a.in + (size_t)((iz * a.H + iy) * a.W + ix) * a.Cs;
    int c0 = kc * 32 + hsel * 16;
    int c1 = c0 + 8;
    g0 = (xv && c0 < a.Cs) ? base + c0 : a.zp;
    g1 = (xv && c1 < a.Cs) ? base + c1 : a.zp;
  };

  v8f acc[NPW];
  #pragma unroll
  for (int j = 0; j < NPW; ++j) { v8f z = {}; acc[j] = z; }

#if HAS_ASYNC_LDS
  // 4 waves x 2 buffers x 512 halves (1 KiB) = 8 KiB LDS double buffer.
  __shared__ u16 sA[4 * 2 * 512];
  auto issueA = [&](int it, int buf) {
    const u16 *g0, *g1;
    srcPtrs(it, g0, g1);
    u16* l = &sA[wave * 1024 + buf * 512 + lane * 16];
    async_ld16(g0, l);
    async_ld16(g1, l + 8);
  };
  issueA(0, 0);
#endif

  #pragma unroll 1
  for (int it = 0; it < ITERS; ++it) {
    const int tap = it >> lgKC;
    const int kc  = it & (KC - 1);
#if HAS_ASYNC_LDS
    if (it + 1 < ITERS) { issueA(it + 1, (it + 1) & 1); wait_async<2>(); }
    else                { wait_async<0>(); }
    const u16* l = &sA[wave * 1024 + (it & 1) * 512 + lane * 16];
    u32x4 lo = *(const u32x4*)(l);
    u32x4 hi = *(const u32x4*)(l + 8);
#else
    const u16 *g0, *g1;
    srcPtrs(it, g0, g1);
    u32x4 lo = *(const u32x4*)g0;
    u32x4 hi = *(const u32x4*)g1;
#endif
    u32x8 au = { lo.x, lo.y, lo.z, lo.w, hi.x, hi.y, hi.z, hi.w };
    v16bf afr = __builtin_bit_cast(v16bf, au);
    // B fragments for this (tap,kc): NPW chunks, 512 halves apart per nc.
    const u16* wbase = a.wpk + ((((size_t)(tap * KC + kc)) << a.lgNC) << 9) + lane * 16;
    #pragma unroll
    for (int j = 0; j < NPW; ++j) {
      const u16* wp = wbase + ((size_t)(ncg * NPW + j) << 9);
      v16bf bfr = __builtin_bit_cast(v16bf, *(const u32x8*)wp);
      acc[j] = __builtin_amdgcn_wmma_f32_16x16x32_bf16(
          false, afr, false, bfr, (short)0, acc[j], false, false);
    }
  }

  // Epilogue. C layout: lanes 0-15 -> N=lane, M=r ; lanes 16-31 -> N=lane-16, M=r+8.
  const int N = lane & 15;
  const int mrow0 = hsel * 8;
  #pragma unroll
  for (int j = 0; j < NPW; ++j) {
    const int n = ((ncg * NPW + j) << 4) + N;
    const float bv = a.bias[n];
    const float gv = (a.flags & F_BN) ? a.gam[n] : 1.0f;
    const float bb = (a.flags & F_BN) ? a.bet[n] : 0.0f;
    #pragma unroll
    for (int r = 0; r < 8; ++r) {
      int oxr = xt + r + mrow0;
      if (oxr >= a.OW) continue;
      u32 oidx = (u32)((oz * a.OH + oy) * a.OW + oxr);
      float v = acc[j][r] + bv;
      if (a.flags & F_BN)   v = v * gv + bb;
      if (a.flags & F_RELU) v = fmaxf(v, 0.0f);
      if (a.flags & F_RES)  v = fmaxf(v + bf2f(a.resid[(oidx << lgCout) + n]), 0.0f);
      if (a.flags & F_MASK) v *= a.mask[oidx];
      if (a.flags & F_OUTF) a.outf[(oidx << lgCout) + n] = v;
      else                  a.out [(oidx << lgCout) + n] = f2bf(v);
    }
  }
}

// ---------------------------------------------------------------------
// 3x3(x3) max-pool of the occupancy mask, SAME padding, arbitrary stride.
// ---------------------------------------------------------------------
__global__ void k_maskpool(const float* __restrict__ m, float* __restrict__ mo,
                           int D, int H, int W, int OD, int OH, int OW,
                           int sz, int sy, int sx) {
  int id = blockIdx.x * blockDim.x + threadIdx.x;
  if (id >= OD * OH * OW) return;
  int x = id % OW; int y = (id / OW) % OH; int z = id / (OW * OH);
  float v = 0.0f;
  for (int dz = 0; dz < 3; ++dz) {
    int iz = z * sz + dz - 1; if ((unsigned)iz >= (unsigned)D) continue;
    for (int dy = 0; dy < 3; ++dy) {
      int iy = y * sy + dy - 1; if ((unsigned)iy >= (unsigned)H) continue;
      for (int dx = 0; dx < 3; ++dx) {
        int ix = x * sx + dx - 1; if ((unsigned)ix >= (unsigned)W) continue;
        v = fmaxf(v, m[(size_t)(iz * H + iy) * W + ix]);
      }
    }
  }
  mo[id] = v;
}

// ---------------------------------------------------------------------
// BEV multiscale fusion: comb = f3 + place(f4,2) + place(f5,4); z-sum -> bev.
// Also mbev = max_z clip(m3 + place(m4,2) + place(m5,4), 0, 1).
// ---------------------------------------------------------------------
__global__ void k_bev(const u16* __restrict__ f3, const u16* __restrict__ f4,
                      const u16* __restrict__ f5,
                      const float* __restrict__ m3, const float* __restrict__ m4,
                      const float* __restrict__ m5,
                      u16* __restrict__ bev, float* __restrict__ mbev) {
  int id = blockIdx.x * blockDim.x + threadIdx.x;
  if (id >= 32 * 32 * 128) return;
  int c = id & 127; int x = (id >> 7) & 31; int y = id >> 12;
  float s = bf2f(f3[((0 * 32 + y) * 32 + x) * 128 + c])
          + bf2f(f3[((1 * 32 + y) * 32 + x) * 128 + c]);
  if (((y | x) & 1) == 0) s += bf2f(f4[((y >> 1) * 16 + (x >> 1)) * 128 + c]);
  if (((y | x) & 3) == 0) s += bf2f(f5[((y >> 2) * 8 + (x >> 2)) * 128 + c]);
  bev[id] = f2bf(s);
  if (c == 0) {
    float mm = 0.0f;
    for (int z = 0; z < 2; ++z) {
      float tt = m3[(z * 32 + y) * 32 + x];
      if (z == 0) {
        if (((y | x) & 1) == 0) tt += m4[(y >> 1) * 16 + (x >> 1)];
        if (((y | x) & 3) == 0) tt += m5[(y >> 2) * 8 + (x >> 2)];
      }
      tt = fminf(fmaxf(tt, 0.0f), 1.0f);
      mm = fmaxf(mm, tt);
    }
    mbev[y * 32 + x] = mm;
  }
}

// ---------------------------------------------------------------------
// GroupNorm(G=32 over [32,32,128]) * g + beta, then exact GELU, in place.
// ---------------------------------------------------------------------
__global__ void k_gn_gelu(float* __restrict__ x, const float* __restrict__ g,
                          const float* __restrict__ be) {
  __shared__ float s1[256], s2[256];
  int grp = blockIdx.x, tid = threadIdx.x;
  float a = 0.0f, b = 0.0f;
  for (int i = tid; i < 4096; i += 256) {
    int hw = i >> 2; int c = grp * 4 + (i & 3);
    float v = x[hw * 128 + c];
    a += v; b += v * v;
  }
  s1[tid] = a; s2[tid] = b; __syncthreads();
  for (int st = 128; st > 0; st >>= 1) {
    if (tid < st) { s1[tid] += s1[tid + st]; s2[tid] += s2[tid + st]; }
    __syncthreads();
  }
  float mu  = s1[0] * (1.0f / 4096.0f);
  float var = s2[0] * (1.0f / 4096.0f) - mu * mu;
  float inv = rsqrtf(var + 1e-5f);
  for (int i = tid; i < 4096; i += 256) {
    int hw = i >> 2; int c = grp * 4 + (i & 3);
    float v = (x[hw * 128 + c] - mu) * inv;
    v = v * g[c] + be[c];
    x[hw * 128 + c] = 0.5f * v * (1.0f + erff(v * 0.70710678118654752f));
  }
}

// 1x1 conv head: [32,32,128] x [128,10] + b -> d_out fp32.
__global__ void k_head(const float* __restrict__ hs, const float* __restrict__ w,
                       const float* __restrict__ b, float* __restrict__ out) {
  int id = blockIdx.x * blockDim.x + threadIdx.x;
  if (id >= 32 * 32 * 10) return;
  int o = id % 10; int hw = id / 10;
  float s = b[o];
  #pragma unroll 4
  for (int c = 0; c < 128; ++c) s += hs[hw * 128 + c] * w[c * 10 + o];
  out[id] = s;
}

// =====================================================================
// Host side
// =====================================================================
static inline int idiv(i64 a, i64 b) { return (int)((a + b - 1) / b); }
static inline int ilog2(int v) { int l = 0; while ((1 << l) < v) ++l; return l; }

struct PP { const float *b, *beta, *g, *w; };

static void run_conv(hipStream_t s, const u16* in, u16* out, float* outf,
                     PP p, const u16* resid, const float* mask, u16* wpk,
                     const u16* zp,
                     int D, int H, int Wd, int OD, int OH, int OW,
                     int sz, int sy, int sx, int CinTrue, int Cs, int Cout,
                     int KZ, int flags) {
  int KC = (Cs + 31) / 32; if (KC < 1) KC = 1;
  int NC = Cout / 16;
  int NPW = (NC >= 4) ? 4 : NC;
  int NCG = NC / NPW;
  int KT = KZ * 9;
  int packN = KT * KC * NC * 32;
  k_pack_w<<<idiv(packN, 256), 256, 0, s>>>(p.w, wpk, KT, KC, NC, CinTrue, Cout);

  int XT = (OW + 15) / 16;
  ConvArgs a;
  a.in = in; a.wpk = wpk; a.zp = zp;
  a.bias = p.b; a.gam = p.g; a.bet = p.beta;
  a.resid = resid; a.mask = mask; a.out = out; a.outf = outf;
  a.D = D; a.H = H; a.W = Wd; a.OD = OD; a.OH = OH; a.OW = OW;
  a.sz = sz; a.sy = sy; a.sx = sx; a.Cs = Cs; a.flags = flags;
  a.lgNC = ilog2(NC); a.lgNCG = ilog2(NCG);
  a.lgXT = ilog2(XT); a.lgOH = ilog2(OH);
  a.totalWaves = OD * OH * XT * NCG;
  int blocks = idiv(a.totalWaves, 4);
  if      (KC == 1 && KZ == 3 && NPW == 1) k_conv_wmma<1, 3, 1><<<blocks, 128, 0, s>>>(a);
  else if (KC == 1 && KZ == 3 && NPW == 2) k_conv_wmma<1, 3, 2><<<blocks, 128, 0, s>>>(a);
  else if (KC == 1 && KZ == 3 && NPW == 4) k_conv_wmma<1, 3, 4><<<blocks, 128, 0, s>>>(a);
  else if (KC == 2 && KZ == 3 && NPW == 4) k_conv_wmma<2, 3, 4><<<blocks, 128, 0, s>>>(a);
  else if (KC == 4 && KZ == 3 && NPW == 4) k_conv_wmma<4, 3, 4><<<blocks, 128, 0, s>>>(a);
  else if (KC == 4 && KZ == 1 && NPW == 4) k_conv_wmma<4, 1, 4><<<blocks, 128, 0, s>>>(a);
  else                                     k_conv_wmma<1, 3, 1><<<blocks, 128, 0, s>>>(a);
}

extern "C" void kernel_launch(void* const* d_in, const int* in_sizes, int n_in,
                              void* d_out, int out_size, void* d_ws, size_t ws_size,
                              hipStream_t stream) {
  (void)n_in; (void)out_size; (void)ws_size;
  const float* vf = (const float*)d_in[0];
  const int*   vc = (const int*)d_in[1];
  const int N = in_sizes[0] / 5;

  // params pytree leaves (jax tree_flatten: dict keys sorted, lists in order),
  // each conv dict flattens as b,beta,g,w:
  //  0 conv_input | 4 conv_out | 8 modality | 12 sem | 16 sem_head{b,w}
  //  18 shared_out | 22 stage0[0] | 26 stage0[1]
  //  30+20*i: stages[i]{blocks[0]{c1,c2}, blocks[1]{c1,c2}, down}
  auto PL = [&](int leaf) -> const float* { return (const float*)d_in[3 + leaf]; };
  auto getp = [&](int base) -> PP { return PP{PL(base), PL(base + 1), PL(base + 2), PL(base + 3)}; };

  const size_t MB = 1ull << 20;
  u8* W = (u8*)d_ws;
  u16* A  = (u16*)(W);
  u16* Bb = (u16*)(W + 64 * MB);
  u16* C  = (u16*)(W + 128 * MB);
  u16* wpk = (u16*)(W + 192 * MB);
  u16* zp  = (u16*)(W + 195 * MB);       // 4 KiB zero page
  float* m0 = (float*)(W + 196 * MB);
  float* m1 = m0 + 16 * 256 * 256;
  float* m2 = m1 + 8 * 128 * 128;
  float* m3 = m2 + 4 * 64 * 64;
  float* m4 = m3 + 2 * 32 * 32;
  float* m5 = m4 + 16 * 16;
  u8* SM = W + 204 * MB;
  u16* f3b = (u16*)SM;
  u16* f4b = f3b + 2 * 32 * 32 * 128;
  u16* f5b = f4b + 16 * 16 * 128;
  u16* bev = f5b + 8 * 8 * 128;
  u16* t1  = bev + 32 * 32 * 128;
  u16* t2  = t1 + 32 * 32 * 128;
  u16* t3  = t2 + 32 * 32 * 128;
  float* hs   = (float*)(t3 + 32 * 32 * 128);
  float* mbev = hs + 32 * 32 * 128;
  float* mo   = mbev + 32 * 32;

  // --- voxelize: zero grid+masks+zero-page, scatter, convert to bf16 ---
  float* grid = (float*)Bb;                              // fp32 [16,256,256,8] in pool B
  (void)hipMemsetAsync(grid, 0, (size_t)16 * 256 * 256 * 8 * 4, stream);
  (void)hipMemsetAsync(zp, 0, 4096, stream);
  size_t maskBytes = (size_t)(16 * 256 * 256 + 8 * 128 * 128 + 4 * 64 * 64 +
                              2 * 32 * 32 + 16 * 16 + 8 * 8) * 4;
  (void)hipMemsetAsync(m0, 0, maskBytes, stream);
  k_scatter<<<idiv(N, 256), 256, 0, stream>>>(vf, vc, N, grid, m0);
  int ge = 16 * 256 * 256 * 8;
  k_cvt_f32_bf16<<<idiv(ge, 256), 256, 0, stream>>>(grid, A, ge);

  // --- conv_input (5->32) + stage0 (32->16->16), submanifold semantics ---
  run_conv(stream, A, Bb, nullptr, getp(0), nullptr, m0, wpk, zp,
           16, 256, 256, 16, 256, 256, 1, 1, 1, 5, 8, 32, 3, F_BN | F_RELU | F_MASK);
  run_conv(stream, Bb, C, nullptr, getp(22), nullptr, m0, wpk, zp,
           16, 256, 256, 16, 256, 256, 1, 1, 1, 32, 32, 16, 3, F_BN | F_RELU | F_MASK);
  run_conv(stream, C, A, nullptr, getp(26), nullptr, m0, wpk, zp,
           16, 256, 256, 16, 256, 256, 1, 1, 1, 16, 16, 16, 3, F_BN | F_RELU | F_MASK);

  u16 *cur = A, *fa = Bb, *fb = C;
  struct SD { int D, H, W; };
  const SD L[6] = {{16,256,256},{8,128,128},{4,64,64},{2,32,32},{1,16,16},{1,8,8}};
  const int CHout[6] = {16, 32, 64, 128, 128, 128};
  const int SS[6][3] = {{0,0,0},{2,2,2},{2,2,2},{2,2,2},{2,2,2},{1,2,2}};
  float* MK[6] = {m0, m1, m2, m3, m4, m5};

  for (int i = 1; i <= 5; ++i) {
    SD di = L[i - 1], dn = L[i];
    int sz = SS[i][0], sy = SS[i][1], sx = SS[i][2];
    int chP = CHout[i - 1], chO = CHout[i];
    int base = 30 + (i - 1) * 20;
    // dilated output mask for the strided SparseConv3d
    k_maskpool<<<idiv((i64)dn.D * dn.H * dn.W, 256), 256, 0, stream>>>(
        MK[i - 1], MK[i], di.D, di.H, di.W, dn.D, dn.H, dn.W, sz, sy, sx);
    // down conv
    run_conv(stream, cur, fa, nullptr, getp(base + 16), nullptr, MK[i], wpk, zp,
             di.D, di.H, di.W, dn.D, dn.H, dn.W, sz, sy, sx,
             chP, chP, chO, 3, F_BN | F_RELU | F_MASK);
    { u16* tmp = cur; cur = fa; fa = tmp; }
    // two residual blocks
    for (int b2 = 0; b2 < 2; ++b2) {
      run_conv(stream, cur, fa, nullptr, getp(base + b2 * 8 + 0), nullptr, MK[i], wpk, zp,
               dn.D, dn.H, dn.W, dn.D, dn.H, dn.W, 1, 1, 1,
               chO, chO, chO, 3, F_BN | F_RELU | F_MASK);
      run_conv(stream, fa, fb, nullptr, getp(base + b2 * 8 + 4), cur, MK[i], wpk, zp,
               dn.D, dn.H, dn.W, dn.D, dn.H, dn.W, 1, 1, 1,
               chO, chO, chO, 3, F_BN | F_RES | F_MASK);
      u16* t0 = cur; cur = fb; fb = fa; fa = t0;
    }
    if (i == 3) (void)hipMemcpyAsync(f3b, cur, (size_t)2 * 32 * 32 * 128 * 2, hipMemcpyDeviceToDevice, stream);
    if (i == 4) (void)hipMemcpyAsync(f4b, cur, (size_t)16 * 16 * 128 * 2, hipMemcpyDeviceToDevice, stream);
    if (i == 5) (void)hipMemcpyAsync(f5b, cur, (size_t)8 * 8 * 128 * 2, hipMemcpyDeviceToDevice, stream);
  }

  // --- BEV fusion + mask dilation ---
  k_bev<<<idiv(32 * 32 * 128, 256), 256, 0, stream>>>(f3b, f4b, f5b, m3, m4, m5, bev, mbev);
  k_maskpool<<<idiv(32 * 32, 256), 256, 0, stream>>>(mbev, mo, 1, 32, 32, 1, 32, 32, 1, 1, 1);

  // --- 2D conv chain at [1,32,32,128] ---
  run_conv(stream, bev, t1, nullptr, getp(4), nullptr, mo, wpk, zp,
           1, 32, 32, 1, 32, 32, 1, 1, 1, 128, 128, 128, 1, F_BN | F_RELU | F_MASK);
  run_conv(stream, t1, t2, nullptr, getp(18), nullptr, mo, wpk, zp,
           1, 32, 32, 1, 32, 32, 1, 1, 1, 128, 128, 128, 1, F_BN | F_RELU | F_MASK);
  run_conv(stream, t2, t3, nullptr, getp(8), nullptr, nullptr, wpk, zp,
           1, 32, 32, 1, 32, 32, 1, 1, 1, 128, 128, 128, 1, F_BN | F_RELU);
  run_conv(stream, t3, nullptr, hs, getp(12), nullptr, nullptr, wpk, zp,
           1, 32, 32, 1, 32, 32, 1, 1, 1, 128, 128, 128, 1, F_OUTF);

  // --- GroupNorm + GELU (uses sem's g/beta), then 1x1 head -> d_out ---
  k_gn_gelu<<<32, 256, 0, stream>>>(hs, PL(12 + 2), PL(12 + 1));
  k_head<<<idiv(32 * 32 * 10, 256), 256, 0, stream>>>(hs, PL(17), PL(16), (float*)d_out);
}
